// DenseQConv1D_84542136255139
// MI455X (gfx1250) — matrix operand order repaired and checked by reading, not compiled, not run on hardware
//
#include <hip/hip_runtime.h>
#include <hip/hip_bf16.h>
#include <math.h>

// ---------------------------------------------------------------------------
// DenseQConv1D on MI455X (gfx1250), wave32 + WMMA.
//
// out[c,n] = s_n^T M_c s_n,  M_c = U_c[:128,:] diag(sign) U_c[:128,:]^T,
// U_c = E @ kron(RY(theta[c,0]),...,RY(theta[c,8])).
//
// ws layout (bytes):
//   [0,            2097152)  S16  : f16, NPAD x 128 row-major normalized patches
//   [2097152,      6291456)  U    : f32, 16 x 128 x 512
//   [6291456,      6815744)  Mfrag: f16, 16 ch x (4 kt x 8 ct x 32 lane x 16) B-frag order
// total ~6.75 MB
// ---------------------------------------------------------------------------

#define B_      8
#define C_IN_   16
#define C_OUT_  16
#define L_      1024
#define KS_     8
#define LOUT_   1017          // L - KS + 1
#define NP_     (B_ * LOUT_)  // 8136 patches
#define NPAD_   8192
#define D_      128           // C_IN * KS
#define DIM_    512

typedef _Float16 v16h __attribute__((ext_vector_type(16)));
typedef _Float16 v8h  __attribute__((ext_vector_type(8)));
typedef float    v8f  __attribute__((ext_vector_type(8)));

// ---------------------------------------------------------------------------
// K1: extract patches, L2-normalize, store f16 row-major S16[NPAD][128].
// One 128-thread block per patch row (pad rows -> zero).
// ---------------------------------------------------------------------------
__global__ __launch_bounds__(128) void build_states(const float* __restrict__ x,
                                                    _Float16* __restrict__ S16) {
    const int n = blockIdx.x;
    const int t = threadIdx.x;
    if (n >= NP_) {                       // zero the padding rows
        S16[n * D_ + t] = (_Float16)0.0f;
        return;
    }
    const int b = n / LOUT_;
    const int l = n % LOUT_;
    const int ci = t >> 3;                // d = ci*8 + k
    const int k  = t & 7;
    const float v = x[(b * C_IN_ + ci) * L_ + l + k];

    __shared__ float red[128];
    red[t] = v * v;
    __syncthreads();
    #pragma unroll
    for (int s = 64; s > 0; s >>= 1) {
        if (t < s) red[t] += red[t + s];
        __syncthreads();
    }
    const float norm = fmaxf(sqrtf(red[0]), 1e-12f);
    S16[n * D_ + t] = (_Float16)(v / norm);
}

// ---------------------------------------------------------------------------
// K2: U[c][d][e] = sum_j E[d,j] R_c[j,e]  (d < 128 only).
// Computed as w = R_c^T * (row d of E) via 9 sequential 2x2 qubit butterflies
// (kron factors commute) -> 9*512 FMAs per row instead of a 512x512 dot.
// grid = (128 d, 16 c), 128 threads.
// ---------------------------------------------------------------------------
__global__ __launch_bounds__(128) void build_u(const float* __restrict__ E,
                                               const float* __restrict__ theta,
                                               float* __restrict__ U) {
    const int d = blockIdx.x;
    const int c = blockIdx.y;
    const int t = threadIdx.x;

    __shared__ float v[DIM_];
    __shared__ float rc[9], rs[9];
    if (t < 9) {
        const float th = 0.5f * theta[c * 9 + t];
        rc[t] = cosf(th);
        rs[t] = sinf(th);
    }
    for (int i = t; i < DIM_; i += 128) v[i] = E[d * DIM_ + i];
    __syncthreads();

    // qubit q acts on bit (8-q) (qubit 0 = MSB). r^T pair op:
    //   w0 =  c*v0 + s*v1 ;  w1 = -s*v0 + c*v1
    for (int q = 0; q < 9; q++) {
        const int stride = 1 << (8 - q);
        const float cq = rc[q], sq = rs[q];
        #pragma unroll
        for (int p = t; p < DIM_ / 2; p += 128) {
            const int i0 = ((p & ~(stride - 1)) << 1) | (p & (stride - 1));
            const int i1 = i0 + stride;
            const float v0 = v[i0], v1 = v[i1];
            v[i0] = cq * v0 + sq * v1;
            v[i1] = cq * v1 - sq * v0;
        }
        __syncthreads();
    }
    for (int i = t; i < DIM_; i += 128)
        U[((c * D_ + d) << 9) + i] = v[i];
}

// ---------------------------------------------------------------------------
// K3: M_c[d][d'] = sum_e sign(e) U[c][d][e] U[c][d'][e], sign = +1 (e<256), -1.
// 32x32 output tile per block, e-chunks of 128 staged in LDS (stride 129 to
// avoid bank conflicts). Result written as f16 directly in WMMA B-fragment
// lane order: Mfrag[c][kt][ct][lane][slot],
//   lane = (klocal>=16)*16 + (d'&15), slot = klocal&15, kt = d>>5, ct = d'>>4.
// grid = (16 tiles, 16 c), 256 threads.
// ---------------------------------------------------------------------------
__global__ __launch_bounds__(256) void build_m(const float* __restrict__ U,
                                               _Float16* __restrict__ Mfrag) {
    const int tt = blockIdx.x;
    const int c  = blockIdx.y;
    const int i0 = (tt >> 2) * 32;
    const int j0 = (tt & 3) * 32;
    const int t  = threadIdx.x;
    const float* __restrict__ Uc = U + c * D_ * DIM_;

    __shared__ float Ua[32][129];
    __shared__ float Ub[32][129];
    float acc[4] = {0.f, 0.f, 0.f, 0.f};

    for (int ech = 0; ech < 4; ech++) {
        const float sgn = (ech < 2) ? 1.0f : -1.0f;
        for (int i = t; i < 32 * 128; i += 256) {
            const int r = i >> 7, e = i & 127;
            Ua[r][e] = Uc[(i0 + r) * DIM_ + ech * 128 + e];
            Ub[r][e] = Uc[(j0 + r) * DIM_ + ech * 128 + e];
        }
        __syncthreads();
        #pragma unroll
        for (int k = 0; k < 4; k++) {
            const int m  = t + k * 256;
            const int ld = m >> 5, lj = m & 31;
            float s = 0.f;
            for (int e = 0; e < 128; e++) s += Ua[ld][e] * Ub[lj][e];
            acc[k] += sgn * s;
        }
        __syncthreads();
    }
    #pragma unroll
    for (int k = 0; k < 4; k++) {
        const int m  = t + k * 256;
        const int d  = i0 + (m >> 5);
        const int dp = j0 + (m & 31);
        const int kt = d >> 5, klocal = d & 31;
        const int lane = ((klocal >> 4) << 4) | (dp & 15);
        const int slot = klocal & 15;
        const int ct   = dp >> 4;
        Mfrag[(((c * 4 + kt) * 8 + ct) * 32 + lane) * 16 + slot] = (_Float16)acc[k];
    }
}

// ---------------------------------------------------------------------------
// K4: per (channel, n-tile): T(16x128) = S_tile(16x128) @ M_c(128x128) via
// 8 col-tiles x 4 k-steps of v_wmma_f32_16x16x32_f16 (f32 accumulate), then
// fused epilogue out[c,n] = sum_d' T[n,d'] * S[n,d'] using the C/D layout
// (lane = col, VGPR = row) + shfl_xor butterfly over each 16-lane half.
// grid = (64 ntile-groups, 16 c), 256 threads = 8 waves, 1 n-tile per wave.
// ---------------------------------------------------------------------------
__global__ __launch_bounds__(256) void qconv_gemm(const _Float16* __restrict__ S16,
                                                  const _Float16* __restrict__ Mfrag,
                                                  float* __restrict__ out) {
    const int c = blockIdx.y;

    __shared__ _Float16 Ml[4 * 8 * 32 * 16];   // 32 KB: this channel's B frags
    {
        const float4* __restrict__ src = (const float4*)(Mfrag + c * 16384);
        float4* dst = (float4*)Ml;
        for (int i = threadIdx.x; i < 2048; i += 256) dst[i] = src[i];
    }
    __syncthreads();

    const int wid   = threadIdx.x >> 5;
    const int lane  = threadIdx.x & 31;
    const int ntile = blockIdx.x * 8 + wid;
    const int nbase = ntile * 16;
    const int h = lane >> 4;       // lane half
    const int r = lane & 15;       // row/col position within half

    // A fragments: lane holds row (nbase+r); K runs {kt*32+h*16..+15}
    const _Float16* __restrict__ row = S16 + (nbase + r) * D_;
    v16h a[4];
    #pragma unroll
    for (int kt = 0; kt < 4; kt++) {
        const v8h lo = *(const v8h*)(row + kt * 32 + h * 16);
        const v8h hi = *(const v8h*)(row + kt * 32 + h * 16 + 8);
        #pragma unroll
        for (int i = 0; i < 8; i++) { a[kt][i] = lo[i]; a[kt][i + 8] = hi[i]; }
    }

    float part[8] = {0.f, 0.f, 0.f, 0.f, 0.f, 0.f, 0.f, 0.f};

    #pragma unroll
    for (int ct = 0; ct < 8; ct++) {
        v8f acc = {};
        #pragma unroll
        for (int kt = 0; kt < 4; kt++) {
            const v16h bfrag = *(const v16h*)(Ml + ((kt * 8 + ct) * 32 + lane) * 16);
            acc = __builtin_amdgcn_wmma_f32_16x16x32_f16(
                false, a[kt], false, bfrag, (short)0, acc, false, false);
        }
        // acc[v] = T[row = h*8+v][col = ct*16+r] ; fuse dot with S
        #pragma unroll
        for (int v = 0; v < 8; v++) {
            const float sv = (float)S16[(nbase + h * 8 + v) * D_ + ct * 16 + r];
            part[v] += acc[v] * sv;
        }
    }

    // reduce across the 16 lanes of each half (cols) -> per-row totals
    #pragma unroll
    for (int v = 0; v < 8; v++) {
        float p = part[v];
        p += __shfl_xor(p, 1, 32);
        p += __shfl_xor(p, 2, 32);
        p += __shfl_xor(p, 4, 32);
        p += __shfl_xor(p, 8, 32);
        part[v] = p;
    }

    if (r == 0) {                   // lanes 0 (rows 0..7) and 16 (rows 8..15)
        #pragma unroll
        for (int v = 0; v < 8; v++) {
            const int n = nbase + h * 8 + v;
            if (n < NP_) {
                const int b = n / LOUT_;
                const int l = n % LOUT_;
                out[(b * C_OUT_ + c) * LOUT_ + l] = part[v];
            }
        }
    }
}

// ---------------------------------------------------------------------------
extern "C" void kernel_launch(void* const* d_in, const int* in_sizes, int n_in,
                              void* d_out, int out_size, void* d_ws, size_t ws_size,
                              hipStream_t stream) {
    (void)in_sizes; (void)n_in; (void)out_size; (void)ws_size;
    const float* x     = (const float*)d_in[0];   // (8,16,1024) f32
    const float* theta = (const float*)d_in[1];   // (16,9)      f32
    const float* E     = (const float*)d_in[2];   // (512,512)   f32
    float* out = (float*)d_out;                   // (8,16,1017) f32

    char* ws = (char*)d_ws;
    _Float16* S16   = (_Float16*)(ws);
    float*    U     = (float*)(ws + (size_t)2097152);
    _Float16* Mfrag = (_Float16*)(ws + (size_t)2097152 + (size_t)4194304);

    build_states<<<NPAD_, 128, 0, stream>>>(x, S16);
    build_u<<<dim3(D_, C_OUT_), 128, 0, stream>>>(E, theta, U);
    build_m<<<dim3(16, C_OUT_), 256, 0, stream>>>(U, Mfrag);
    qconv_gemm<<<dim3(NPAD_ / 128, C_OUT_), 256, 0, stream>>>(S16, Mfrag, out);
}
